// EncoderLayer_79723182949061
// MI455X (gfx1250) — matrix operand order repaired and checked by reading, not compiled, not run on hardware
//
#include <hip/hip_runtime.h>

// ---------------- problem constants (from reference) ----------------
constexpr int CB   = 4;
constexpr int CL   = 2048;
constexpr int CD   = 1024;
constexpr int CH   = 16;
constexpr int CDFF = 4096;
constexpr int CM   = CB * CL;          // 8192 rows

// ---------------- WMMA types ----------------
typedef __attribute__((ext_vector_type(16))) __bf16 v16bf;
typedef __attribute__((ext_vector_type(8)))  float  v8f;

union FragU  { uint4 u[2]; v16bf v; __bf16 e[16]; };

__device__ __forceinline__ v16bf make_frag(const __bf16* p0, const __bf16* p1) {
    FragU f;
    f.u[0] = *(const uint4*)p0;
    f.u[1] = *(const uint4*)p1;
    return f.v;
}

__device__ __forceinline__ v8f wmma_bf16(v16bf a, v16bf b, v8f c) {
    return __builtin_amdgcn_wmma_f32_16x16x32_bf16(false, a, false, b,
                                                   (short)0, c, false, false);
}

__device__ __forceinline__ void lds_fence() {
#if __has_builtin(__builtin_amdgcn_s_wait_dscnt)
    __builtin_amdgcn_s_wait_dscnt(0);
#endif
}

// wave32 half-reductions (lanes 0..15 / 16..31 independently)
__device__ __forceinline__ float rmax16(float v) {
#pragma unroll
    for (int m = 1; m < 16; m <<= 1) v = fmaxf(v, __shfl_xor(v, m, 32));
    return v;
}
__device__ __forceinline__ float rsum16(float v) {
#pragma unroll
    for (int m = 1; m < 16; m <<= 1) v += __shfl_xor(v, m, 32);
    return v;
}

// ---------------- Tensor Data Mover staging ----------------
#if __has_builtin(__builtin_amdgcn_tensor_load_to_lds) && \
    __has_builtin(__builtin_amdgcn_s_wait_tensorcnt)
#define USE_TDM 1
#else
#define USE_TDM 0
#endif

#if USE_TDM
typedef unsigned int u32x4 __attribute__((ext_vector_type(4)));
typedef int          i32x4 __attribute__((ext_vector_type(4)));
typedef int          i32x8 __attribute__((ext_vector_type(8)));
typedef __attribute__((address_space(3))) const void lds_cv;

__device__ __forceinline__ unsigned lds_byte_off(const void* p) {
    return (unsigned)(uintptr_t)(lds_cv*)p;     // addrspacecast -> LDS offset
}

// DMA one 128-row x 64-elem bf16 tile (row stride kStride elems) into LDS,
// padding each 64-elem row with 8 elems (LDS pitch 72 elems = 144B).
__device__ __forceinline__ void tdm_load_tile(const __bf16* gsrc, unsigned ldsoff,
                                              unsigned kStride) {
    unsigned long long ga = (unsigned long long)(uintptr_t)gsrc;
    u32x4 g0;
    g0.x = 1u;                                            // count=1 (valid user D#)
    g0.y = ldsoff;                                        // lds_addr (bytes)
    g0.z = (unsigned)ga;                                  // global_addr[31:0]
    g0.w = (unsigned)((ga >> 32) & 0x1FFFFFFu) | (2u << 30); // addr[56:32], type=2
    i32x8 g1;
    g1[0] = (1 << 16)        // data_size = 2 bytes
          | (1 << 20)        // pad_enable
          | (4 << 22)        // pad_interval = 32 DWORDs (one 64-elem row)
          | (3 << 25);       // pad_amount  = 4 DWORDs (8 bf16)
    g1[1] = (int)(64u  << 16);   // tensor_dim0 = 64   (bits 79:48)
    g1[2] = (int)(128u << 16);   // tensor_dim1 = 128  (bits 111:80)
    g1[3] = (int)(64u  << 16);   // tile_dim0 = 64     (bits 127:112)
    g1[4] = 128;                 // tile_dim1 = 128    (bits 143:128)
    g1[5] = (int)kStride;        // tensor_dim0_stride (bits 207:160)
    g1[6] = 0;
    g1[7] = 0;
    i32x4 gz = {0, 0, 0, 0};
#if __clang_major__ >= 23
    i32x8 gz8 = {0, 0, 0, 0, 0, 0, 0, 0};
    __builtin_amdgcn_tensor_load_to_lds(g0, g1, gz, gz, gz8, 0);
#else
    __builtin_amdgcn_tensor_load_to_lds(g0, g1, gz, gz, 0);
#endif
}
#endif // USE_TDM

// ---------------- f32 -> bf16 conversion (vectorized) ----------------
__global__ __launch_bounds__(256) void cvt_bf16_kernel(const float* __restrict__ s,
                                                       __bf16* __restrict__ d, int n) {
    int i4 = blockIdx.x * 256 + threadIdx.x;
    if (i4 * 4 < n) {
        float4 f = ((const float4*)s)[i4];
        union { __bf16 e[4]; uint2 u; } o;
        o.e[0] = (__bf16)f.x; o.e[1] = (__bf16)f.y;
        o.e[2] = (__bf16)f.z; o.e[3] = (__bf16)f.w;
        ((uint2*)d)[i4] = o.u;
    }
}

// f32 [K][N] -> bf16 [N][K] transpose-convert (one-time pass over weights)
__global__ __launch_bounds__(256)
void cvt_transpose_kernel(const float* __restrict__ s, __bf16* __restrict__ d,
                          int K, int N) {
    __shared__ float tile[32][33];
    const int k0 = blockIdx.y * 32, n0 = blockIdx.x * 32;
    const int tx = threadIdx.x & 31, ty = threadIdx.x >> 5;
#pragma unroll
    for (int j = 0; j < 4; ++j) {
        int r = ty + j * 8;
        tile[r][tx] = s[(size_t)(k0 + r) * N + n0 + tx];
    }
    __syncthreads();
#pragma unroll
    for (int j = 0; j < 4; ++j) {
        int r = ty + j * 8;
        d[(size_t)(n0 + r) * K + k0 + tx] = (__bf16)tile[tx][r];
    }
}

// mask (bool) -> additive bias (0 or -1e12)
__global__ __launch_bounds__(256) void maskbias_kernel(const unsigned char* __restrict__ m,
                                                       float* __restrict__ o, int n) {
    int i = blockIdx.x * 256 + threadIdx.x;
    if (i < n) o[i] = m[i] ? -1.0e12f : 0.0f;
}

// ---------------- bf16 WMMA GEMM, TDM-staged, double-buffered ----------------
// C[M,N] = A[M,K] * Bt[N,K]^T + bias ; optional relu.
// Block tile 128x128, K-step 64 (two WMMA sub-steps, 16 WMMAs per stage).
// LDS pitch 72 elems (64 data + 8 pad) to match the TDM pad engine.
__global__ __launch_bounds__(256)
void gemm_bf16_kernel(const __bf16* __restrict__ A, const __bf16* __restrict__ Bt,
                      const float* __restrict__ bias,
                      float* __restrict__ outF, __bf16* __restrict__ outB,
                      __bf16* __restrict__ outBT,
                      int M, int N, int K, int relu) {
    __shared__ __align__(16) __bf16 As[2][128 * 72];
    __shared__ __align__(16) __bf16 Bs[2][128 * 72];

    const int t    = threadIdx.x;
    const int lane = t & 31;
    const int w    = t >> 5;
    const int wm   = w & 3;
    const int wn   = w >> 2;
    const int l15  = lane & 15;
    const int hi   = lane >> 4;
    const int m0   = blockIdx.y * 128;
    const int n0   = blockIdx.x * 128;

    v8f zero = {};
    v8f acc[2][4];
#pragma unroll
    for (int i = 0; i < 2; ++i)
#pragma unroll
        for (int j = 0; j < 4; ++j) acc[i][j] = zero;

    // --- staging helper: fill buffer `buf` with the K-tile starting at k0 ---
    auto stage = [&](int buf, int k0) {
#if USE_TDM
        if (w == 0) {   // one wave issues both tile DMAs (EXEC ignored by TDM)
            tdm_load_tile(A  + (size_t)m0 * K + k0, lds_byte_off(&As[buf][0]), (unsigned)K);
            tdm_load_tile(Bt + (size_t)n0 * K + k0, lds_byte_off(&Bs[buf][0]), (unsigned)K);
        }
#else
        const int srow = t >> 1;           // 0..127
        const int scol = (t & 1) * 32;     // 0 or 32 elems
        const __bf16* ag = A  + (size_t)(m0 + srow) * K + k0 + scol;
        const __bf16* bg = Bt + (size_t)(n0 + srow) * K + k0 + scol;
        uint4 a0 = *(const uint4*)ag,       a1 = *(const uint4*)(ag + 8);
        uint4 a2 = *(const uint4*)(ag + 16), a3 = *(const uint4*)(ag + 24);
        uint4 b0 = *(const uint4*)bg,       b1 = *(const uint4*)(bg + 8);
        uint4 b2 = *(const uint4*)(bg + 16), b3 = *(const uint4*)(bg + 24);
        __bf16* as = &As[buf][srow * 72 + scol];
        __bf16* bs = &Bs[buf][srow * 72 + scol];
        *(uint4*)(as)      = a0; *(uint4*)(as + 8)  = a1;
        *(uint4*)(as + 16) = a2; *(uint4*)(as + 24) = a3;
        *(uint4*)(bs)      = b0; *(uint4*)(bs + 8)  = b1;
        *(uint4*)(bs + 16) = b2; *(uint4*)(bs + 24) = b3;
        __builtin_prefetch(ag + 64, 0, 1);
        __builtin_prefetch(bg + 64, 0, 1);
#endif
    };

    const int nK = K / 64;
    stage(0, 0);
    int cur = 0;
    for (int i = 0; i < nK; ++i) {
#if USE_TDM
        if (w == 0) __builtin_amdgcn_s_wait_tensorcnt(0);
#endif
        __syncthreads();                       // cur buffer ready for all waves
        if (i + 1 < nK) stage(cur ^ 1, (i + 1) * 64);   // overlap DMA with WMMA

#pragma unroll
        for (int s = 0; s < 2; ++s) {          // two K=32 sub-steps
            v16bf afr[2];
#pragma unroll
            for (int mt = 0; mt < 2; ++mt) {
                int r = wm * 32 + mt * 16 + l15;
                const __bf16* p = &As[cur][r * 72 + s * 32 + (hi ? 8 : 0)];
                afr[mt] = make_frag(p, p + 16);
            }
            v16bf bfr[4];
#pragma unroll
            for (int nt = 0; nt < 4; ++nt) {
                int c = wn * 64 + nt * 16 + l15;
                const __bf16* p = &Bs[cur][c * 72 + s * 32 + (hi ? 8 : 0)];
                bfr[nt] = make_frag(p, p + 16);
            }
#pragma unroll
            for (int mt = 0; mt < 2; ++mt)
#pragma unroll
                for (int nt = 0; nt < 4; ++nt)
                    acc[mt][nt] = wmma_bf16(afr[mt], bfr[nt], acc[mt][nt]);
        }
        cur ^= 1;
    }

    // ---- epilogue ----
#pragma unroll
    for (int nt = 0; nt < 4; ++nt) {
        int col = n0 + wn * 64 + nt * 16 + l15;
        float bv = bias ? bias[col] : 0.0f;
#pragma unroll
        for (int mt = 0; mt < 2; ++mt) {
#pragma unroll
            for (int i = 0; i < 8; ++i) {
                int row = m0 + wm * 32 + mt * 16 + i + hi * 8;
                float v = acc[mt][nt][i] + bv;
                if (relu) v = fmaxf(v, 0.0f);
                if (outF)  outF[(size_t)row * N + col] = v;
                if (outB)  outB[(size_t)row * N + col] = (__bf16)v;
                if (outBT) outBT[(size_t)col * M + row] = (__bf16)v;
            }
        }
    }
}

// ---------------- flash attention (barrier-free, bf16 WMMA) ----------------
__global__ __launch_bounds__(256)
void attn_kernel(const __bf16* __restrict__ Q, const __bf16* __restrict__ Km,
                 const __bf16* __restrict__ Vt, const float* __restrict__ mbias,
                 __bf16* __restrict__ O) {
    __shared__ __align__(16) __bf16 Pt[8][16 * 40];

    const int t    = threadIdx.x;
    const int lane = t & 31;
    const int w    = t >> 5;
    const int l15  = lane & 15;
    const int hi   = lane >> 4;
    const int bh   = blockIdx.y;
    const int b    = bh >> 4;
    const int h    = bh & 15;
    const int q0   = blockIdx.x * 128 + w * 16;
    const float SCALE = 0.125f;

    v16bf qf[2];
    {
        const size_t qbase = (size_t)(b * CL + q0 + l15) * CD + h * 64;
#pragma unroll
        for (int s = 0; s < 2; ++s) {
            const __bf16* p = Q + qbase + s * 32 + (hi ? 8 : 0);
            qf[s] = make_frag(p, p + 16);
        }
    }

    v8f zero = {};
    v8f oacc[4] = {zero, zero, zero, zero};
    float m_run[8], l_run[8];
#pragma unroll
    for (int i = 0; i < 8; ++i) { m_run[i] = -1.0e30f; l_run[i] = 0.0f; }

    for (int kb = 0; kb < CL / 32; ++kb) {
        v8f sv[2] = {zero, zero};
#pragma unroll
        for (int nt = 0; nt < 2; ++nt) {
            int krow = kb * 32 + nt * 16 + l15;
            const size_t kbase = (size_t)(b * CL + krow) * CD + h * 64;
#pragma unroll
            for (int s = 0; s < 2; ++s) {
                const __bf16* kp = Km + kbase + s * 32 + (hi ? 8 : 0);
                v16bf kf = make_frag(kp, kp + 16);
                sv[nt] = wmma_bf16(qf[s], kf, sv[nt]);
            }
        }

        float mb0 = mbias[b * CL + kb * 32 + l15];
        float mb1 = mbias[b * CL + kb * 32 + 16 + l15];
#pragma unroll
        for (int i = 0; i < 8; ++i) {
            float s0 = sv[0][i] * SCALE + mb0;
            float s1 = sv[1][i] * SCALE + mb1;
            float mx   = rmax16(fmaxf(s0, s1));
            float mnew = fmaxf(m_run[i], mx);
            float corr = __expf(m_run[i] - mnew);
            float p0 = __expf(s0 - mnew);
            float p1 = __expf(s1 - mnew);
            l_run[i] = l_run[i] * corr + rsum16(p0 + p1);
            m_run[i] = mnew;
#pragma unroll
            for (int tt = 0; tt < 4; ++tt) oacc[tt][i] *= corr;
            int prow = i + hi * 8;
            Pt[w][prow * 40 + l15]      = (__bf16)p0;
            Pt[w][prow * 40 + 16 + l15] = (__bf16)p1;
        }
        lds_fence();

        const __bf16* pp = &Pt[w][l15 * 40 + (hi ? 8 : 0)];
        v16bf pf = make_frag(pp, pp + 16);
#pragma unroll
        for (int nt2 = 0; nt2 < 4; ++nt2) {
            const __bf16* vp = Vt + (size_t)(h * 64 + nt2 * 16 + l15) * CM
                                  + (b * CL + kb * 32 + (hi ? 8 : 0));
            v16bf vf = make_frag(vp, vp + 16);
            oacc[nt2] = wmma_bf16(pf, vf, oacc[nt2]);
        }
    }

    float inv[8];
#pragma unroll
    for (int i = 0; i < 8; ++i) inv[i] = 1.0f / l_run[i];
#pragma unroll
    for (int nt2 = 0; nt2 < 4; ++nt2) {
#pragma unroll
        for (int i = 0; i < 8; ++i) {
            int row = q0 + i + hi * 8;
            O[(size_t)(b * CL + row) * CD + h * 64 + nt2 * 16 + l15] =
                (__bf16)(oacc[nt2][i] * inv[i]);
        }
    }
}

// ---------------- fused residual-add + LayerNorm ----------------
__global__ __launch_bounds__(256)
void add_ln_kernel(const float* __restrict__ x, const float* __restrict__ r,
                   const float* __restrict__ g, const float* __restrict__ be,
                   float* __restrict__ outF, __bf16* __restrict__ outB) {
    __shared__ float sb[256];
    const int t = threadIdx.x;
    const size_t row = blockIdx.x;
    const float* xr = x + row * CD;
    const float* rr = r + row * CD;
    float v[4];
    float s = 0.0f;
#pragma unroll
    for (int j = 0; j < 4; ++j) { int c = t + j * 256; v[j] = xr[c] + rr[c]; s += v[j]; }
    sb[t] = s; __syncthreads();
    for (int o = 128; o > 0; o >>= 1) { if (t < o) sb[t] += sb[t + o]; __syncthreads(); }
    float mean = sb[0] * (1.0f / CD);
    __syncthreads();
    float s2 = 0.0f;
#pragma unroll
    for (int j = 0; j < 4; ++j) { float d = v[j] - mean; s2 += d * d; }
    sb[t] = s2; __syncthreads();
    for (int o = 128; o > 0; o >>= 1) { if (t < o) sb[t] += sb[t + o]; __syncthreads(); }
    float rstd = rsqrtf(sb[0] * (1.0f / CD) + 1e-5f);
#pragma unroll
    for (int j = 0; j < 4; ++j) {
        int c = t + j * 256;
        float o = (v[j] - mean) * rstd * g[c] + be[c];
        outF[row * CD + c] = o;
        if (outB) outB[row * CD + c] = (__bf16)o;
    }
}

// ---------------- orchestration ----------------
extern "C" void kernel_launch(void* const* d_in, const int* in_sizes, int n_in,
                              void* d_out, int out_size, void* d_ws, size_t ws_size,
                              hipStream_t stream) {
    (void)in_sizes; (void)n_in; (void)out_size; (void)ws_size;
    const float* x    = (const float*)d_in[0];
    const unsigned char* mask = (const unsigned char*)d_in[1];
    const float* wq = (const float*)d_in[2];  const float* bq = (const float*)d_in[3];
    const float* wk = (const float*)d_in[4];  const float* bk = (const float*)d_in[5];
    const float* wv = (const float*)d_in[6];  const float* bv = (const float*)d_in[7];
    const float* wo = (const float*)d_in[8];  const float* bo = (const float*)d_in[9];
    const float* w1 = (const float*)d_in[10]; const float* b1 = (const float*)d_in[11];
    const float* w2 = (const float*)d_in[12]; const float* b2 = (const float*)d_in[13];
    const float* g1 = (const float*)d_in[14]; const float* be1 = (const float*)d_in[15];
    const float* g2 = (const float*)d_in[16]; const float* be2 = (const float*)d_in[17];

    const size_t MiB = 1ull << 20;
    char* W = (char*)d_ws;
    __bf16* xb   = (__bf16*)(W + 0);
    __bf16* qb   = (__bf16*)(W + 16 * MiB);
    __bf16* kbuf = (__bf16*)(W + 32 * MiB);
    __bf16* vbt  = (__bf16*)(W + 48 * MiB);    // V^T [CD][CM] bf16
    __bf16* hb   = (__bf16*)(W + 0);           // 64MiB alias (xb..vbt dead by then)
    __bf16* wqt  = (__bf16*)(W + 64 * MiB);    // all weights TRANSPOSED [N][K]
    __bf16* wkt  = (__bf16*)(W + 66 * MiB);
    __bf16* wvt  = (__bf16*)(W + 68 * MiB);
    __bf16* wot  = (__bf16*)(W + 70 * MiB);
    __bf16* w1t  = (__bf16*)(W + 72 * MiB);
    __bf16* w2t  = (__bf16*)(W + 80 * MiB);
    float*  mbia = (float*) (W + 88 * MiB);
    __bf16* ob   = (__bf16*)(W + 89 * MiB);
    float*  ap   = (float*) (W + 105 * MiB);   // attn proj, later reused as ff
    float*  y1   = (float*) (W + 137 * MiB);
    __bf16* y1b  = (__bf16*)(W + 169 * MiB);

    cvt_bf16_kernel<<<dim3((CM * CD / 4 + 255) / 256), 256, 0, stream>>>(x, xb, CM * CD);
    dim3 t1024(CD / 32, CD / 32);
    cvt_transpose_kernel<<<t1024, 256, 0, stream>>>(wq, wqt, CD, CD);
    cvt_transpose_kernel<<<t1024, 256, 0, stream>>>(wk, wkt, CD, CD);
    cvt_transpose_kernel<<<t1024, 256, 0, stream>>>(wv, wvt, CD, CD);
    cvt_transpose_kernel<<<t1024, 256, 0, stream>>>(wo, wot, CD, CD);
    cvt_transpose_kernel<<<dim3(CDFF / 32, CD / 32), 256, 0, stream>>>(w1, w1t, CD, CDFF);
    cvt_transpose_kernel<<<dim3(CD / 32, CDFF / 32), 256, 0, stream>>>(w2, w2t, CDFF, CD);
    maskbias_kernel<<<dim3((CB * CL + 255) / 256), 256, 0, stream>>>(mask, mbia, CB * CL);

    dim3 g1024(CD / 128, CM / 128);
    gemm_bf16_kernel<<<g1024, 256, 0, stream>>>(xb, wqt, bq, nullptr, qb,   nullptr, CM, CD, CD, 0);
    gemm_bf16_kernel<<<g1024, 256, 0, stream>>>(xb, wkt, bk, nullptr, kbuf, nullptr, CM, CD, CD, 0);
    gemm_bf16_kernel<<<g1024, 256, 0, stream>>>(xb, wvt, bv, nullptr, nullptr, vbt, CM, CD, CD, 0);

    attn_kernel<<<dim3(CL / 128, CB * CH), 256, 0, stream>>>(qb, kbuf, vbt, mbia, ob);

    gemm_bf16_kernel<<<g1024, 256, 0, stream>>>(ob, wot, bo, ap, nullptr, nullptr, CM, CD, CD, 0);
    add_ln_kernel<<<dim3(CM), 256, 0, stream>>>(x, ap, g1, be1, y1, y1b);

    gemm_bf16_kernel<<<dim3(CDFF / 128, CM / 128), 256, 0, stream>>>(
        y1b, w1t, b1, nullptr, hb, nullptr, CM, CDFF, CD, 1);
    gemm_bf16_kernel<<<g1024, 256, 0, stream>>>(hb, w2t, b2, ap, nullptr, nullptr, CM, CD, CDFF, 0);

    add_ln_kernel<<<dim3(CM), 256, 0, stream>>>(y1, ap, g2, be2, (float*)d_out, nullptr);
}
